// DDGCRNCell_43104291783474
// MI455X (gfx1250) — compile-verified, matrix operand order checked by reading, not run on hardware
//
#include <hip/hip_runtime.h>
#include <hip/hip_bf16.h>
#include <math.h>

// ---------------- CDNA5 WMMA types ----------------
typedef __attribute__((ext_vector_type(16))) __bf16 v16bf;
typedef __attribute__((ext_vector_type(8)))  float  v8f;

// ---------------- problem constants ----------------
#define NB    64      // batch
#define NN    883     // nodes
#define NPAD  896     // 56 * 16
#define NT    56      // 16-row tiles
#define NCH   28      // 32-row chunks
#define EDIM  10
#define CINW  66      // DIN + DOUT
#define KIN   132     // 2 * CIN (x_g flattened)
#define DOUTW 64

__device__ __forceinline__ int kidx(int j, int half) {
  // element j (0..15) of a lane's 16x16x32 bf16 A/B fragment -> K index (0..31)
  return ((j < 8) ? j : (j + 8)) + (half ? 8 : 0);
}
__device__ __forceinline__ void k_to_jh(int K, int& j, int& half) {
  if (K < 8)       { half = 0; j = K; }
  else if (K < 16) { half = 1; j = K - 8; }
  else if (K < 24) { half = 0; j = K - 8; }
  else             { half = 1; j = K - 16; }
}
__device__ __forceinline__ float sigm(float x) { return 1.0f / (1.0f + __expf(-x)); }

__device__ __forceinline__ v8f wmma_bf16(v16bf a, v16bf b, v8f c) {
  return __builtin_amdgcn_wmma_f32_16x16x32_bf16(false, a, false, b, (short)0, c,
                                                 false, false);
}

// ---------------- workspace layout (bytes, all 256-aligned) ----------------
#define ALIGN256(x) (((x) + 255) & ~((size_t)255))
static const size_t SZ_VEC = ALIGN256((size_t)NB * NT * 512 * 2);            // nodevec packs (bf16)
static const size_t SZ_XSB = ALIGN256((size_t)NB * NCH * 5 * 512 * 2);       // xs B-operand packs
static const size_t SZ_XS  = ALIGN256((size_t)NB * NN * CINW * 4);           // xs f32
static const size_t SZ_Y   = ALIGN256((size_t)NB * NN * CINW * 4);           // y = L@xs f32
static const size_t SZ_D   = ALIGN256((size_t)NB * NPAD * 4);                // d f32
static const size_t SZ_XGA = ALIGN256((size_t)NN * 4 * 5 * 512 * 2);         // x_g A-operand packs
static const size_t SZ_W   = ALIGN256((size_t)NN * 8 * 5 * 512 * 2);         // W B-operand packs (gate max)
static const size_t SZ_ZR  = ALIGN256((size_t)NB * NN * 128 * 4);            // gate z_r f32

// ================= kernel 1: hypernet + nodevec + operand packing =================
__global__ void ddg_prep(const float* __restrict__ x, const float* __restrict__ state,
                         const float* __restrict__ emb, const float* __restrict__ timee,
                         const float* __restrict__ dayy, const float* __restrict__ speed,
                         const float* __restrict__ occupy,
                         const float* __restrict__ fc1w, const float* __restrict__ fc1b,
                         const float* __restrict__ fc2w, const float* __restrict__ fc2b,
                         const float* __restrict__ fc3w, const float* __restrict__ fc3b,
                         const float* __restrict__ zr,
                         float* __restrict__ xs_out, __bf16* __restrict__ vecpack,
                         __bf16* __restrict__ xsbpack, int phase) {
  int idx = blockIdx.x * blockDim.x + threadIdx.x;
  if (idx >= NB * NPAD) return;
  int b = idx / NPAD, n = idx % NPAD;
  int tile = n >> 4, M = n & 15;
  int chunk = n >> 5, Kr = n & 31;
  int jb, hb; k_to_jh(Kr, jb, hb);

  if (n >= NN) {   // padded rows: zero both packs
    __bf16* vp = vecpack + ((size_t)(b * NT + tile)) * 512;
#pragma unroll
    for (int t = 0; t < 16; t++) { vp[M * 16 + t] = (__bf16)0.f; vp[(16 + M) * 16 + t] = (__bf16)0.f; }
    for (int ct = 0; ct < 5; ct++)
      for (int c16 = 0; c16 < 16; c16++)
        xsbpack[(((size_t)(b * NCH + chunk)) * 5 + ct) * 512 + (hb * 16 + c16) * 16 + jb] = (__bf16)0.f;
    return;
  }

  // xs = concat(x, state) (gate) or concat(x, z*state) (update)
  float xs[CINW];
  size_t bn = (size_t)b * NN + n;
  xs[0] = x[bn * 2 + 0];
  xs[1] = x[bn * 2 + 1];
#pragma unroll 8
  for (int c = 0; c < DOUTW; c++) {
    float s = state[bn * DOUTW + c];
    if (phase == 1) s *= zr[bn * 128 + c];   // z = first half of z_r
    xs[2 + c] = s;
  }
  for (int c = 0; c < CINW; c++) xs_out[bn * CINW + c] = xs[c];

  // hypernetwork: 66 -> 16 -> 2 -> 10
  float h1[16];
#pragma unroll
  for (int t = 0; t < 16; t++) {
    float a = fc1b[t];
    for (int c = 0; c < CINW; c++) a += xs[c] * fc1w[c * 16 + t];
    h1[t] = sigm(a);
  }
  float h2[2];
#pragma unroll
  for (int u = 0; u < 2; u++) {
    float a = fc2b[u];
    for (int t = 0; t < 16; t++) a += h1[t] * fc2w[t * 2 + u];
    h2[u] = sigm(a);
  }
  float vec[EDIM];
#pragma unroll
  for (int e = 0; e < EDIM; e++) {
    float filt = h2[0] * fc3w[e] + h2[1] * fc3w[EDIM + e] + fc3b[e];
    float dyn = emb[n * EDIM + e] * timee[bn * EDIM + e] * dayy[bn * EDIM + e] *
                speed[bn * EDIM + e] * occupy[bn * EDIM + e];
    vec[e] = tanhf(dyn * filt);
  }

  // nodevec fragment pack (A layout == B layout for V*V^T), K padded 10 -> 32
  __bf16* vp = vecpack + ((size_t)(b * NT + tile)) * 512;
#pragma unroll
  for (int j = 0; j < 16; j++) {
    int k0 = kidx(j, 0), k1 = kidx(j, 1);
    vp[M * 16 + j]        = (__bf16)((k0 < EDIM) ? vec[k0] : 0.f);
    vp[(16 + M) * 16 + j] = (__bf16)((k1 < EDIM) ? vec[k1] : 0.f);
  }

  // xs B-operand pack: B[K=n%32][col=c] over 5 column tiles (66 -> 80 pad)
  for (int ct = 0; ct < 5; ct++)
    for (int c16 = 0; c16 < 16; c16++) {
      int c = ct * 16 + c16;
      float v = (c < CINW) ? xs[c] : 0.f;
      xsbpack[(((size_t)(b * NCH + chunk)) * 5 + ct) * 512 + (hb * 16 + c16) * 16 + jb] = (__bf16)v;
    }
}

// ================= kernel 2: d = rowsum(relu(V V^T))^-1/2 =================
__global__ void __launch_bounds__(32) ddg_dcalc(const __bf16* __restrict__ vecpack,
                                                float* __restrict__ d) {
  int tile = blockIdx.x, b = blockIdx.y, lane = threadIdx.x;
  const __bf16* base = vecpack + (size_t)b * NT * 512;
  v16bf a = *reinterpret_cast<const v16bf*>(base + (size_t)tile * 512 + lane * 16);
  float sum[8];
#pragma unroll
  for (int r = 0; r < 8; r++) sum[r] = 0.f;

  for (int mt = 0; mt < NT; mt++) {
    if (mt + 1 < NT) __builtin_prefetch(base + (size_t)(mt + 1) * 512 + lane * 16, 0, 3);
    v16bf bb = *reinterpret_cast<const v16bf*>(base + (size_t)mt * 512 + lane * 16);
    v8f c = {};
    c = wmma_bf16(a, bb, c);
#pragma unroll
    for (int r = 0; r < 8; r++) sum[r] += fmaxf(c[r], 0.f);
  }
  // reduce 16 columns held across the 16 lanes of each half
#pragma unroll
  for (int m = 1; m < 16; m <<= 1)
#pragma unroll
    for (int r = 0; r < 8; r++) sum[r] += __shfl_xor(sum[r], m, 32);

  int half = lane >> 4;
  if ((lane & 15) == 0) {
#pragma unroll
    for (int r = 0; r < 8; r++) {
      int n = tile * 16 + half * 8 + r;
      d[(size_t)b * NPAD + n] = (n < NN) ? rsqrtf(sum[r]) : 0.f;  // pad -> 0 (kills 0*inf)
    }
  }
}

// ================= kernel 3: y = xs - (relu(VV^T) .* d d^T) @ xs =================
__global__ void __launch_bounds__(32) ddg_lap(const __bf16* __restrict__ vecpack,
                                              const __bf16* __restrict__ xsbpack,
                                              const float* __restrict__ d,
                                              const float* __restrict__ xs_f32,
                                              float* __restrict__ y) {
  __shared__ __bf16 S[16 * 32];  // scaled adjacency sub-block, row-major [row][K]
  int tile = blockIdx.x, b = blockIdx.y, lane = threadIdx.x;
  int half = lane >> 4, c16 = lane & 15;
  const __bf16* vbase = vecpack + (size_t)b * NT * 512;
  v16bf a = *reinterpret_cast<const v16bf*>(vbase + (size_t)tile * 512 + lane * 16);

  float dn[8];
#pragma unroll
  for (int r = 0; r < 8; r++) dn[r] = d[(size_t)b * NPAD + tile * 16 + half * 8 + r];

  v8f acc[5] = {};
  for (int mc = 0; mc < NCH; mc++) {
#pragma unroll
    for (int sub = 0; sub < 2; sub++) {
      int mt = mc * 2 + sub;
      v16bf bb = *reinterpret_cast<const v16bf*>(vbase + (size_t)mt * 512 + lane * 16);
      v8f c = {};
      c = wmma_bf16(a, bb, c);
      float dm = d[(size_t)b * NPAD + mt * 16 + c16];
#pragma unroll
      for (int r = 0; r < 8; r++) {
        float s = fmaxf(c[r], 0.f) * dn[r] * dm;
        S[(half * 8 + r) * 32 + sub * 16 + c16] = (__bf16)s;
      }
    }
    __syncthreads();
    // re-gather S as an A-operand fragment (row = lane%16, K via kidx)
    v16bf sa;
    const __bf16* srow = &S[(lane & 15) * 32];
#pragma unroll
    for (int j = 0; j < 16; j++) sa[j] = srow[kidx(j, half)];
#pragma unroll
    for (int ct = 0; ct < 5; ct++) {
      v16bf xb = *reinterpret_cast<const v16bf*>(
          xsbpack + (((size_t)(b * NCH + mc)) * 5 + ct) * 512 + lane * 16);
      acc[ct] = wmma_bf16(sa, xb, acc[ct]);
    }
    __syncthreads();
  }
#pragma unroll
  for (int ct = 0; ct < 5; ct++)
#pragma unroll
    for (int r = 0; r < 8; r++) {
      int n = tile * 16 + half * 8 + r;
      int c = ct * 16 + c16;
      if (n < NN && c < CINW) {
        size_t o = ((size_t)b * NN + n) * CINW + c;
        y[o] = xs_f32[o] - acc[ct][r];
      }
    }
}

// ================= kernel 4a: per-node W pack (W_n = sum_e emb * wpool) =================
__global__ void ddg_wpack(const float* __restrict__ emb, const float* __restrict__ wpool,
                          __bf16* __restrict__ wpk, int O, int CT) {
  int n = blockIdx.x, ct = blockIdx.y;
  float em[EDIM];
#pragma unroll
  for (int e = 0; e < EDIM; e++) em[e] = emb[n * EDIM + e];
  for (int el = threadIdx.x; el < 5 * 512; el += 256) {
    int kc = el / 512, rem = el % 512;
    int lane = rem >> 4, j = rem & 15;
    int half = lane >> 4, col = lane & 15;
    int ki = kc * 32 + kidx(j, half);
    int o = ct * 16 + col;
    float v = 0.f;
    if (ki < KIN) {
      int k = ki / CINW, i = ki % CINW;
      const float* wp = wpool + ((size_t)k * CINW + i) * O + o;
      size_t estr = (size_t)2 * CINW * O;
#pragma unroll
      for (int e = 0; e < EDIM; e++) v += em[e] * wp[e * estr];
    }
    wpk[(((size_t)n * CT + ct) * 5 + kc) * 512 + rem] = (__bf16)v;
  }
}

// ================= kernel 4b: x_g A-operand pack =================
__global__ void ddg_xgapack(const float* __restrict__ xs, const float* __restrict__ y,
                            __bf16* __restrict__ xga) {
  size_t idx = (size_t)blockIdx.x * 256 + threadIdx.x;
  size_t total = (size_t)NN * 4 * 5 * 512;
  if (idx >= total) return;
  int rem = (int)(idx & 511);
  size_t t = idx >> 9;
  int kc = (int)(t % 5); t /= 5;
  int rt = (int)(t % 4);
  int n  = (int)(t / 4);
  int lane = rem >> 4, j = rem & 15;
  int half = lane >> 4;
  int bidx = rt * 16 + (lane & 15);           // A row = batch index
  int ki = kc * 32 + kidx(j, half);
  float v = 0.f;
  if (ki < CINW)      v = xs[((size_t)bidx * NN + n) * CINW + ki];
  else if (ki < KIN)  v = y[((size_t)bidx * NN + n) * CINW + (ki - CINW)];
  xga[idx] = (__bf16)v;
}

// ================= kernel 5: per-node GEMM + bias + nonlinearity / GRU =================
__global__ void __launch_bounds__(32) ddg_gemm(const __bf16* __restrict__ xga,
                                               const __bf16* __restrict__ wpk,
                                               const float* __restrict__ emb,
                                               const float* __restrict__ bpool,
                                               const float* __restrict__ state,
                                               const float* __restrict__ zr_in,
                                               float* __restrict__ out, int O, int CT,
                                               int phase) {
  int n = blockIdx.x;
  int tt = blockIdx.y;
  int ct = tt % CT, rt = tt / CT;
  int lane = threadIdx.x;
  v8f acc = {};
#pragma unroll
  for (int kc = 0; kc < 5; kc++) {
    v16bf a = *reinterpret_cast<const v16bf*>(
        xga + (((size_t)n * 4 + rt) * 5 + kc) * 512 + lane * 16);
    v16bf w = *reinterpret_cast<const v16bf*>(
        wpk + (((size_t)n * CT + ct) * 5 + kc) * 512 + lane * 16);
    acc = wmma_bf16(a, w, acc);
  }
  int half = lane >> 4, col = lane & 15;
  int o = ct * 16 + col;
  float bias = 0.f;
#pragma unroll
  for (int e = 0; e < EDIM; e++) bias += emb[n * EDIM + e] * bpool[e * O + o];
#pragma unroll
  for (int r = 0; r < 8; r++) {
    int b = rt * 16 + half * 8 + r;
    float v = acc[r] + bias;
    size_t bn = (size_t)b * NN + n;
    if (phase == 0) {
      out[bn * 128 + o] = sigm(v);                       // z_r
    } else {
      float hc = tanhf(v);
      float rr = zr_in[bn * 128 + 64 + o];               // r = second half
      float st = state[bn * DOUTW + o];
      out[bn * DOUTW + o] = rr * st + (1.f - rr) * hc;   // final GRU output
    }
  }
}

// ================= launch =================
extern "C" void kernel_launch(void* const* d_in, const int* in_sizes, int n_in,
                              void* d_out, int out_size, void* d_ws, size_t ws_size,
                              hipStream_t stream) {
  (void)in_sizes; (void)n_in; (void)out_size; (void)ws_size;
  const float* x      = (const float*)d_in[0];
  const float* state  = (const float*)d_in[1];
  const float* emb    = (const float*)d_in[2];
  const float* timee  = (const float*)d_in[3];
  const float* dayy   = (const float*)d_in[4];
  const float* speed  = (const float*)d_in[5];
  const float* occupy = (const float*)d_in[6];
  const float* g_wpool = (const float*)d_in[7];
  const float* g_bpool = (const float*)d_in[8];
  const float* g_fc1w = (const float*)d_in[9];  const float* g_fc1b = (const float*)d_in[10];
  const float* g_fc2w = (const float*)d_in[11]; const float* g_fc2b = (const float*)d_in[12];
  const float* g_fc3w = (const float*)d_in[13]; const float* g_fc3b = (const float*)d_in[14];
  const float* u_wpool = (const float*)d_in[15];
  const float* u_bpool = (const float*)d_in[16];
  const float* u_fc1w = (const float*)d_in[17]; const float* u_fc1b = (const float*)d_in[18];
  const float* u_fc2w = (const float*)d_in[19]; const float* u_fc2b = (const float*)d_in[20];
  const float* u_fc3w = (const float*)d_in[21]; const float* u_fc3b = (const float*)d_in[22];

  char* ws = (char*)d_ws;
  size_t off = 0;
  __bf16* vecpack = (__bf16*)(ws + off); off += SZ_VEC;
  __bf16* xsbpack = (__bf16*)(ws + off); off += SZ_XSB;
  float*  xs_f32  = (float*)(ws + off);  off += SZ_XS;
  float*  y_f32   = (float*)(ws + off);  off += SZ_Y;
  float*  d_vec   = (float*)(ws + off);  off += SZ_D;
  __bf16* xga     = (__bf16*)(ws + off); off += SZ_XGA;
  __bf16* wpk     = (__bf16*)(ws + off); off += SZ_W;
  float*  zr      = (float*)(ws + off);  off += SZ_ZR;

  dim3 blkPrep(256), grdPrep((NB * NPAD + 255) / 256);
  dim3 blkWave(32), grdTile(NT, NB);
  dim3 blkW(256);
  int nXgaBlocks = (int)(((size_t)NN * 4 * 5 * 512 + 255) / 256);

  // -------- phase 0: gate DGCN --------
  ddg_prep<<<grdPrep, blkPrep, 0, stream>>>(x, state, emb, timee, dayy, speed, occupy,
      g_fc1w, g_fc1b, g_fc2w, g_fc2b, g_fc3w, g_fc3b, zr,
      xs_f32, vecpack, xsbpack, 0);
  ddg_dcalc<<<grdTile, blkWave, 0, stream>>>(vecpack, d_vec);
  ddg_lap<<<grdTile, blkWave, 0, stream>>>(vecpack, xsbpack, d_vec, xs_f32, y_f32);
  ddg_wpack<<<dim3(NN, 8), blkW, 0, stream>>>(emb, g_wpool, wpk, 128, 8);
  ddg_xgapack<<<nXgaBlocks, 256, 0, stream>>>(xs_f32, y_f32, xga);
  ddg_gemm<<<dim3(NN, 32), blkWave, 0, stream>>>(xga, wpk, emb, g_bpool, state, zr,
                                                 zr, 128, 8, 0);
  // -------- phase 1: update DGCN + GRU combine --------
  ddg_prep<<<grdPrep, blkPrep, 0, stream>>>(x, state, emb, timee, dayy, speed, occupy,
      u_fc1w, u_fc1b, u_fc2w, u_fc2b, u_fc3w, u_fc3b, zr,
      xs_f32, vecpack, xsbpack, 1);
  ddg_dcalc<<<grdTile, blkWave, 0, stream>>>(vecpack, d_vec);
  ddg_lap<<<grdTile, blkWave, 0, stream>>>(vecpack, xsbpack, d_vec, xs_f32, y_f32);
  ddg_wpack<<<dim3(NN, 4), blkW, 0, stream>>>(emb, u_wpool, wpk, 64, 4);
  ddg_xgapack<<<nXgaBlocks, 256, 0, stream>>>(xs_f32, y_f32, xga);
  ddg_gemm<<<dim3(NN, 16), blkWave, 0, stream>>>(xga, wpk, emb, u_bpool, state, zr,
                                                 (float*)d_out, 64, 4, 1);
}
